// ScriptGreedyDecoder_4913442587025
// MI455X (gfx1250) — matrix operand order repaired and checked by reading, not compiled, not run on hardware
//
#include <hip/hip_runtime.h>
#include <hip/hip_bf16.h>
#include <stdint.h>

// ---------------- problem constants ----------------
#define VOCAB   4096
#define ENCD    1024
#define PREDH   640
#define JOINTH  640
#define TFRAMES 128
#define MAXSYM  30
#define SSTEPS  (TFRAMES * (MAXSYM + 1))   // 3968

// ---------------- launch config ----------------
#define NBLK 128        // persistent blocks; must be co-resident (MI455X has >=128 WGPs, tiny LDS/regs here)
#define TPB  256        // 8 wave32 waves per block

typedef __bf16 bf16_t;
typedef __attribute__((ext_vector_type(16))) __bf16 v16bf;
typedef __attribute__((ext_vector_type(8)))  float  v8f;

// ---------------- workspace layout (bytes) ----------------
static constexpr size_t SZ_PKL   = (size_t)2560 * 1280 * 2;          // 6,553,600
static constexpr size_t OFF_PK0  = 0;
static constexpr size_t OFF_PK1  = OFF_PK0 + SZ_PKL;
static constexpr size_t OFF_PKJ1 = OFF_PK1 + SZ_PKL;                 // 13,107,200
static constexpr size_t OFF_PKJ2 = OFF_PKJ1 + (size_t)640 * 1664 * 2;
static constexpr size_t OFF_F    = OFF_PKJ2 + (size_t)4096 * 640 * 2; // 8x640 f32 state vectors
static constexpr size_t OFF_VIN0 = OFF_F    + (size_t)8 * 640 * 4;
static constexpr size_t OFF_VIN1 = OFF_VIN0 + 1280 * 2;
static constexpr size_t OFF_VINJ = OFF_VIN1 + 1280 * 2;
static constexpr size_t OFF_HIDB = OFF_VINJ + 1664 * 2;
static constexpr size_t OFF_MISC = OFF_HIDB + 640 * 2;  // key(8) | state int[5] | barrier uint[2]

// ---------------- helpers ----------------
__device__ __forceinline__ bf16_t f2bf(float f) {
  unsigned u = __float_as_uint(f);
  unsigned r = (u + 0x7FFFu + ((u >> 16) & 1u)) >> 16;   // round-to-nearest-even
  unsigned short s = (unsigned short)r;
  return __builtin_bit_cast(bf16_t, s);
}
__device__ __forceinline__ float sigm(float x) { return 1.0f / (1.0f + expf(-x)); }

// A matrix = GEMV input vector broadcast into row 0 (contiguous k order, matches packed B):
// lane0: elems0-15 = K[0..15]; lane16: K[16..31]; other lanes zero.
__device__ __forceinline__ v16bf load_a(const bf16_t* __restrict__ v, int kbase, int lane) {
  bf16_t z = __builtin_bit_cast(bf16_t, (unsigned short)0);
  v16bf a;
#pragma unroll
  for (int i = 0; i < 16; ++i) a[i] = z;
  if ((lane & 15) == 0) {
    const bf16_t* p = v + kbase + ((lane >> 4) << 4);
#pragma unroll
    for (int i = 0; i < 16; ++i) a[i] = p[i];
  }
  return a;
}

// B tile: packed so each lane loads 16 contiguous bf16 (32B aligned).
__device__ __forceinline__ v16bf load_b(const bf16_t* __restrict__ P, long tile, int lane) {
  return *reinterpret_cast<const v16bf*>(P + (tile << 9) + (lane << 4));
}

#define WMMA_BF16(A, B, C) \
  __builtin_amdgcn_wmma_f32_16x16x32_bf16(false, (A), false, (B), (short)0, (C), false, false)

// ---------------- grid barrier (sense via generation counter) ----------------
__device__ __forceinline__ void gbar(unsigned* cnt, unsigned* gen) {
  __threadfence();
  __syncthreads();
  if (threadIdx.x == 0) {
    unsigned g = ((volatile unsigned*)gen)[0];
    if (atomicAdd(cnt, 1u) == (unsigned)(NBLK - 1)) {
      ((volatile unsigned*)cnt)[0] = 0u;
      __threadfence();
      atomicAdd(gen, 1u);
    } else {
      while (((volatile unsigned*)gen)[0] == g) __builtin_amdgcn_s_sleep(2);
    }
    __threadfence();
  }
  __syncthreads();
}

// ---------------- weight pack: W[N,K] f32 row-major -> bf16 WMMA B-tiles ----------------
// Tile (rt,kt) is 512 bf16; element for lane L, e: W[rt*16+(L&15)][kt*32+(L>>4)*16+e].
__global__ void pack_kernel(const float* __restrict__ W, bf16_t* __restrict__ P,
                            int N, int KT_total, int kofs, int Ksrc) {
  long idx = (long)blockIdx.x * blockDim.x + threadIdx.x;
  if (idx >= (long)N * Ksrc) return;
  int n = (int)(idx / Ksrc), k = (int)(idx % Ksrc);
  int kg = kofs + k;
  int rt = n >> 4, col = n & 15;
  int kt = kg >> 5, kl = kg & 31;
  int lane = ((kl >> 4) << 4) + col;
  int e = kl & 15;
  size_t off = (((size_t)rt * KT_total + kt) << 9) + (size_t)lane * 16 + e;
  P[off] = f2bf(W[(size_t)n * Ksrc + k]);
}

// ---------------- init ----------------
__global__ void init_kernel(float* fz, bf16_t* vin0, bf16_t* vin1, bf16_t* vinJ, bf16_t* hidb,
                            const float* __restrict__ enc, int* state,
                            unsigned long long* key, unsigned* bar, int* labels) {
  int i = blockIdx.x * blockDim.x + threadIdx.x;
  bf16_t z = __builtin_bit_cast(bf16_t, (unsigned short)0);
  if (i < 8 * 640) fz[i] = 0.0f;               // h0,c0,h1,c1,h0n,c0n,h1n,c1n
  if (i < 1280) { vin0[i] = z; vin1[i] = z; }  // x=0 (SOS), h0=0, h1=0
  if (i < 1024) vinJ[i] = f2bf(enc[i]);        // f = enc[t=0]
  else if (i < 1664) vinJ[i] = z;
  if (i < 640) hidb[i] = z;
  if (i < SSTEPS) labels[i] = -1;
  if (i == 0) {
    state[0] = 0; state[1] = 0; state[2] = -1; state[3] = 0; state[4] = 0; // t,sa,last,cnt,done
    *key = 0ull; bar[0] = 0u; bar[1] = 0u;
  }
}

// ---------------- fused LSTM layer: z = Wpk @ vin + bias, gates, cell update ----------------
__device__ __forceinline__ void lstm_phase(
    const bf16_t* __restrict__ PK, const bf16_t* __restrict__ vin,
    const float* __restrict__ bias, const float* __restrict__ cprev,
    float* __restrict__ hn, float* __restrict__ cn, bf16_t* __restrict__ hbf,
    float* smem, int lane, int wv) {
  const int b = blockIdx.x;                      // h-tile (0..39)
  v8f ai = {}; v8f afv = {}; v8f ag = {}; v8f ao = {};
  for (int kt = wv; kt < 40; kt += 8) {
    v16bf a  = load_a(vin, kt * 32, lane);
    v16bf bi = load_b(PK, (long)( b        * 40 + kt), lane);
    v16bf bfv= load_b(PK, (long)((b + 40)  * 40 + kt), lane);
    v16bf bg = load_b(PK, (long)((b + 80)  * 40 + kt), lane);
    v16bf bo = load_b(PK, (long)((b + 120) * 40 + kt), lane);
    ai  = WMMA_BF16(a, bi,  ai);
    afv = WMMA_BF16(a, bfv, afv);
    ag  = WMMA_BF16(a, bg,  ag);
    ao  = WMMA_BF16(a, bo,  ao);
  }
  if (lane < 16) {                                // D row0 = lanes 0-15 of acc[0]
    smem[wv * 64 +  0 + lane] = ai[0];
    smem[wv * 64 + 16 + lane] = afv[0];
    smem[wv * 64 + 32 + lane] = ag[0];
    smem[wv * 64 + 48 + lane] = ao[0];
  }
  __syncthreads();
  if (threadIdx.x < 16) {
    int j = b * 16 + threadIdx.x;
    float zi = bias[j], zf = bias[640 + j], zg = bias[1280 + j], zo = bias[1920 + j];
    for (int w = 0; w < 8; ++w) {
      zi += smem[w * 64 + threadIdx.x];
      zf += smem[w * 64 + 16 + threadIdx.x];
      zg += smem[w * 64 + 32 + threadIdx.x];
      zo += smem[w * 64 + 48 + threadIdx.x];
    }
    float c2 = sigm(zf) * cprev[j] + sigm(zi) * tanhf(zg);
    float hh = sigm(zo) * tanhf(c2);
    hn[j] = hh; cn[j] = c2; hbf[j] = f2bf(hh);
  }
}

// ---------------- persistent greedy decoder ----------------
__global__ __launch_bounds__(TPB) void rnnt_decode(
    const float* __restrict__ enc, const int* __restrict__ outlen,
    const float* __restrict__ emb,
    const float* __restrict__ b0v, const float* __restrict__ b1v,
    const float* __restrict__ bj1, const float* __restrict__ bj2,
    const bf16_t* __restrict__ PK0, const bf16_t* __restrict__ PK1,
    const bf16_t* __restrict__ PKJ1, const bf16_t* __restrict__ PKJ2,
    bf16_t* vin0, bf16_t* vin1, bf16_t* vinJ, bf16_t* hidb,
    float* h0, float* c0, float* h1, float* c1,
    float* h0n, float* c0n, float* h1n, float* c1n,
    int* state, unsigned long long* keyp, unsigned* bcnt, unsigned* bgen,
    float* out_logits, int* out_labels, int* out_cnt) {
  __shared__ float smem[512];
  __shared__ int ssh[4];
  const int lane = threadIdx.x & 31;
  const int wv   = threadIdx.x >> 5;
  const int timesteps = outlen[0];

  for (int s = 0; s < SSTEPS; ++s) {
    int dn = ((volatile int*)state)[4];
    if (dn) {  // decode finished: out_logits row must be zero (where(done,0,logits))
      int gt = blockIdx.x * TPB + threadIdx.x;
      if (gt < VOCAB) out_logits[(size_t)s * VOCAB + gt] = 0.0f;
      gbar(bcnt, bgen);
      continue;
    }

    // ---- P1: LSTM layer 0 on vin0=[x;h0]; writes h0n,c0n and bf16 h0n into vin1[0:640]
    if (blockIdx.x < 40)
      lstm_phase(PK0, vin0, b0v, c0, h0n, c0n, vin1, smem, lane, wv);
    gbar(bcnt, bgen);

    // ---- P2: LSTM layer 1 on vin1=[h0n;h1]; writes h1n,c1n and bf16 h1n into vinJ[1024:]
    if (blockIdx.x < 40)
      lstm_phase(PK1, vin1, b1v, c1, h1n, c1n, vinJ + 1024, smem, lane, wv);
    gbar(bcnt, bgen);

    // ---- PJ1: hid = relu(Wj1 @ [f;h1n] + bj1) -> hidb bf16
    if (blockIdx.x < 40) {
      v8f acc = {};
      for (int kt = wv; kt < 52; kt += 8) {
        v16bf a  = load_a(vinJ, kt * 32, lane);
        v16bf bm = load_b(PKJ1, (long)blockIdx.x * 52 + kt, lane);
        acc = WMMA_BF16(a, bm, acc);
      }
      if (lane < 16) smem[wv * 16 + lane] = acc[0];
      __syncthreads();
      if (threadIdx.x < 16) {
        int n = blockIdx.x * 16 + threadIdx.x;
        float u = bj1[n];
        for (int w = 0; w < 8; ++w) u += smem[w * 16 + threadIdx.x];
        hidb[n] = f2bf(u > 0.0f ? u : 0.0f);
      }
    }
    gbar(bcnt, bgen);

    // ---- PJ2: logits = Wj2 @ hid + bj2; write out row; packed-argmax atomic
    {
      v8f acc = {};
      int rt = blockIdx.x * 2 + (wv >> 2);      // 256 row-tiles, 2 per block, 4 waves each
      for (int kt = (wv & 3); kt < 20; kt += 4) {
        v16bf a  = load_a(hidb, kt * 32, lane);
        v16bf bm = load_b(PKJ2, (long)rt * 20 + kt, lane);
        acc = WMMA_BF16(a, bm, acc);
      }
      if (lane < 16) smem[wv * 16 + lane] = acc[0];
      __syncthreads();
      if (threadIdx.x < 32) {                    // wave 0: 32 logits (two 16-col tiles)
        int half = threadIdx.x >> 4, col = threadIdx.x & 15;
        int n = (blockIdx.x * 2 + half) * 16 + col;
        float v = bj2[n];
        for (int w = 0; w < 4; ++w) v += smem[(half * 4 + w) * 16 + col];
        out_logits[(size_t)s * VOCAB + n] = v;
        unsigned fb = __float_as_uint(v);
        unsigned ob = (fb & 0x80000000u) ? ~fb : (fb | 0x80000000u);  // order-preserving
        unsigned long long key =
            ((unsigned long long)ob << 32) | (unsigned)(VOCAB - 1 - n); // tie -> lowest idx
        for (int off = 16; off > 0; off >>= 1) {
          unsigned lo = (unsigned)key, hi = (unsigned)(key >> 32);
          unsigned olo = __shfl_xor(lo, off, 32);
          unsigned ohi = __shfl_xor(hi, off, 32);
          unsigned long long ok = ((unsigned long long)ohi << 32) | olo;
          if (ok > key) key = ok;
        }
        if (threadIdx.x == 0) atomicMax(keyp, key);
      }
    }
    gbar(bcnt, bgen);

    // ---- Update: argmax decode, scalar state, commit h/c, rebuild vin0/vin1/vinJ
    if (blockIdx.x == 0) {
      if (threadIdx.x == 0) {
        int t = state[0], sa = state[1], last = state[2], cnt = state[3], done2 = state[4];
        unsigned long long key = *keyp;
        int k = (VOCAB - 1) - (int)(unsigned)(key & 0xFFFFFFFFull);
        int stop = (k == 0) || (sa >= MAXSYM);
        int emit = !stop;
        if (emit) { last = k; out_labels[cnt] = k; cnt++; sa++; }
        else      { sa = 0; t++; if (t >= timesteps) done2 = 1; }
        *keyp = 0ull;
        state[0] = t; state[1] = sa; state[2] = last; state[3] = cnt; state[4] = done2;
        out_cnt[0] = cnt;
        ssh[0] = emit; ssh[1] = last; ssh[2] = t;
      }
      __syncthreads();
      int emit = ssh[0], last = ssh[1], t = ssh[2];
      if (emit) {
        for (int j = threadIdx.x; j < 640; j += TPB) {
          h0[j] = h0n[j]; c0[j] = c0n[j]; h1[j] = h1n[j]; c1[j] = c1n[j];
        }
      }
      for (int j = threadIdx.x; j < 640; j += TPB) {
        float x = (last < 0) ? 0.0f : emb[(size_t)last * PREDH + j];
        vin0[j]        = f2bf(x);
        vin0[640 + j]  = f2bf(h0[j]);
        vin1[640 + j]  = f2bf(h1[j]);
      }
      int tf = (t < TFRAMES - 1) ? t : (TFRAMES - 1);
      for (int j = threadIdx.x; j < ENCD; j += TPB)
        vinJ[j] = f2bf(enc[(size_t)tf * ENCD + j]);
    }
    gbar(bcnt, bgen);
  }
}

// ---------------- host ----------------
extern "C" void kernel_launch(void* const* d_in, const int* in_sizes, int n_in,
                              void* d_out, int out_size, void* d_ws, size_t ws_size,
                              hipStream_t stream) {
  (void)in_sizes; (void)n_in; (void)out_size; (void)ws_size;
  const float* enc  = (const float*)d_in[0];
  const int*   olen = (const int*)d_in[1];
  const float* emb  = (const float*)d_in[2];
  const float* Wih0 = (const float*)d_in[3];
  const float* Whh0 = (const float*)d_in[4];
  const float* b0v  = (const float*)d_in[5];
  const float* Wih1 = (const float*)d_in[6];
  const float* Whh1 = (const float*)d_in[7];
  const float* b1v  = (const float*)d_in[8];
  const float* Wj1  = (const float*)d_in[9];
  const float* bj1  = (const float*)d_in[10];
  const float* Wj2  = (const float*)d_in[11];
  const float* bj2  = (const float*)d_in[12];

  char* ws = (char*)d_ws;
  bf16_t* PK0  = (bf16_t*)(ws + OFF_PK0);
  bf16_t* PK1  = (bf16_t*)(ws + OFF_PK1);
  bf16_t* PKJ1 = (bf16_t*)(ws + OFF_PKJ1);
  bf16_t* PKJ2 = (bf16_t*)(ws + OFF_PKJ2);
  float*  fb   = (float*)(ws + OFF_F);
  float *h0 = fb, *c0 = fb + 640, *h1 = fb + 1280, *c1 = fb + 1920;
  float *h0n = fb + 2560, *c0n = fb + 3200, *h1n = fb + 3840, *c1n = fb + 4480;
  bf16_t* vin0 = (bf16_t*)(ws + OFF_VIN0);
  bf16_t* vin1 = (bf16_t*)(ws + OFF_VIN1);
  bf16_t* vinJ = (bf16_t*)(ws + OFF_VINJ);
  bf16_t* hidb = (bf16_t*)(ws + OFF_HIDB);
  unsigned long long* keyp = (unsigned long long*)(ws + OFF_MISC);
  int*      state = (int*)(ws + OFF_MISC + 8);
  unsigned* bar   = (unsigned*)(ws + OFF_MISC + 8 + 20);

  float* out_logits = (float*)d_out;
  int*   out_labels = (int*)(out_logits + (size_t)SSTEPS * VOCAB);
  int*   out_cnt    = out_labels + SSTEPS;

  auto nb = [](long n) { return (int)((n + 255) / 256); };
  // LSTM0: W0 = [Wih0 | Whh0] over K=1280 (40 k-tiles)
  pack_kernel<<<nb(2560L * 640), 256, 0, stream>>>(Wih0, PK0, 2560, 40, 0,   640);
  pack_kernel<<<nb(2560L * 640), 256, 0, stream>>>(Whh0, PK0, 2560, 40, 640, 640);
  // LSTM1
  pack_kernel<<<nb(2560L * 640), 256, 0, stream>>>(Wih1, PK1, 2560, 40, 0,   640);
  pack_kernel<<<nb(2560L * 640), 256, 0, stream>>>(Whh1, PK1, 2560, 40, 640, 640);
  // Joint
  pack_kernel<<<nb(640L * 1664), 256, 0, stream>>>(Wj1, PKJ1, 640,  52, 0, 1664);
  pack_kernel<<<nb(4096L * 640), 256, 0, stream>>>(Wj2, PKJ2, 4096, 20, 0, 640);

  init_kernel<<<20, 256, 0, stream>>>(fb, vin0, vin1, vinJ, hidb, enc, state, keyp, bar, out_labels);

  rnnt_decode<<<NBLK, TPB, 0, stream>>>(
      enc, olen, emb, b0v, b1v, bj1, bj2,
      PK0, PK1, PKJ1, PKJ2,
      vin0, vin1, vinJ, hidb,
      h0, c0, h1, c1, h0n, c0n, h1n, c1n,
      state, keyp, bar, bar + 1,
      out_logits, out_labels, out_cnt);
}
